// TransformerLM_2637109920376
// MI455X (gfx1250) — matrix operand order repaired and checked by reading, not compiled, not run on hardware
//
#include <hip/hip_runtime.h>

// ---------------- problem constants ----------------
#define E_DIM   1024
#define FF_DIM  4096
#define HS      64
#define NH      16
#define SEQ     1024
#define BATCH   4
#define NLAYER  8
#define VOCAB   32000
#define MTOT    (BATCH * SEQ)   // 4096 token rows

typedef __bf16 bf16;
typedef __attribute__((ext_vector_type(16))) __bf16 v16bf;
typedef __attribute__((ext_vector_type(8)))  __bf16 v8bf;
typedef __attribute__((ext_vector_type(8)))  float  v8f;

// ---------------- small helpers ----------------
__device__ inline bf16 f2bf(float f) {
  unsigned u = __builtin_bit_cast(unsigned, f);
  unsigned r = (u + 0x7FFFu + ((u >> 16) & 1u)) >> 16;
  return __builtin_bit_cast(bf16, (unsigned short)(r & 0xFFFFu));
}

// round-to-nearest-even bf16 pair packed into one dword: lo = a, hi = b
__device__ inline unsigned pack2bf(float a, float b) {
  unsigned ua = __builtin_bit_cast(unsigned, a);
  unsigned ub = __builtin_bit_cast(unsigned, b);
  ua = (ua + 0x7FFFu + ((ua >> 16) & 1u)) >> 16;
  ub = ub + 0x7FFFu + ((ub >> 16) & 1u);
  return (ua & 0xFFFFu) | (ub & 0xFFFF0000u);
}

__device__ inline v8f v8f_zero() {
  v8f z = {0.f, 0.f, 0.f, 0.f, 0.f, 0.f, 0.f, 0.f};
  return z;
}

// gfx1250 async global->LDS copy: 16B per lane, tracked by ASYNCcnt.
// VDST = per-lane LDS byte offset (generic LDS pointer truncated to 32 bits,
// ISA 10.2: LDS aperture address = addr[31:0]); VADDR = 64-bit global address.
__device__ inline void async_cp16(bf16* ldst, const bf16* gsrc) {
  unsigned lds = (unsigned)(unsigned long long)(void*)ldst;
  asm volatile("global_load_async_to_lds_b128 %0, %1, off"
               :: "v"(lds), "v"(gsrc) : "memory");
}
__device__ inline void wait_async0() {
  asm volatile("s_wait_asynccnt 0x0" ::: "memory");
}

__device__ inline v16bf frag_cat(v8bf lo, v8bf hi) {
  v16bf r;
#pragma unroll
  for (int i = 0; i < 8; ++i) { r[i] = lo[i]; r[i + 8] = hi[i]; }
  return r;
}

// A-matrix fragment (16x32 bf16).  Per ISA 7.12.2:
// lane m=lane&15, g=lane>>4; VGPR0..3 hold K = kb+g*8+0..7, VGPR4..7 hold K = kb+16+g*8+0..7
__device__ inline v16bf load_frag_a(const bf16* base, int ld, int m0, int kb) {
  int lane = threadIdx.x & 31;
  int m = lane & 15, g = lane >> 4;
  const bf16* p = base + (m0 + m) * ld + kb + g * 8;
  v8bf lo = *(const v8bf*)p;
  v8bf hi = *(const v8bf*)(p + 16);
  return frag_cat(lo, hi);
}

// B-matrix fragment (32x16 bf16) from an LDS tile stored transposed [n][k].
// lane n=lane&15, g=lane>>4; VGPR0..3 hold K = kb+g*8+0..7, VGPR4..7 hold K = kb+16+g*8+0..7
__device__ inline v16bf load_frag_b(const bf16* base, int ld, int n0, int kb) {
  int lane = threadIdx.x & 31;
  int n = lane & 15, g = lane >> 4;
  const bf16* p = base + (n0 + n) * ld + kb + g * 8;
  v8bf lo = *(const v8bf*)p;
  v8bf hi = *(const v8bf*)(p + 16);
  return frag_cat(lo, hi);
}

__device__ inline v8f wmma_bf16(v16bf a, v16bf b, v8f c) {
  return __builtin_amdgcn_wmma_f32_16x16x32_bf16(false, a, false, b, (short)0, c,
                                                 false, false);
}

// ---------------- embedding: z = tok_emb[x] + pos_emb ----------------
__global__ __launch_bounds__(256)
void embed_kernel(const int* __restrict__ x, const float* __restrict__ tok,
                  const float* __restrict__ pos, float* __restrict__ z) {
  const long row = blockIdx.x;                 // 0..MTOT-1
  const int t = (int)(row & (SEQ - 1));
  const long tokoff = (long)x[row] * E_DIM;
  for (int i = threadIdx.x; i < E_DIM; i += 256)
    z[row * E_DIM + i] = tok[tokoff + i] + pos[(long)t * E_DIM + i];
}

// ---------------- layernorm (f32 in, bf16 out) ----------------
__global__ __launch_bounds__(256)
void layernorm_kernel(const float* __restrict__ z, const float* __restrict__ gamma,
                      const float* __restrict__ beta, bf16* __restrict__ out) {
  __shared__ float red[256];
  const int tid = threadIdx.x;
  const long row = blockIdx.x;
  const float* x = z + row * E_DIM;

  float s = 0.f;
  for (int i = tid; i < E_DIM; i += 256) s += x[i];
  red[tid] = s;
  __syncthreads();
  for (int off = 128; off > 0; off >>= 1) {
    if (tid < off) red[tid] += red[tid + off];
    __syncthreads();
  }
  float mean = red[0] * (1.f / E_DIM);
  __syncthreads();

  float s2 = 0.f;
  for (int i = tid; i < E_DIM; i += 256) { float d = x[i] - mean; s2 += d * d; }
  red[tid] = s2;
  __syncthreads();
  for (int off = 128; off > 0; off >>= 1) {
    if (tid < off) red[tid] += red[tid + off];
    __syncthreads();
  }
  float rstd = rsqrtf(red[0] * (1.f / E_DIM) + 1e-5f);

  bf16* o = out + row * E_DIM;
  for (int i = tid; i < E_DIM; i += 256)
    o[i] = f2bf((x[i] - mean) * rstd * gamma[i] + beta[i]);
}

// ---------------- f32 -> bf16 convert (packed stores) ----------------
__global__ __launch_bounds__(256)
void cvt_f32_bf16_kernel(const float* __restrict__ in, bf16* __restrict__ out, int n2) {
  int i = blockIdx.x * 256 + threadIdx.x;     // one bf16 PAIR per thread
  if (i < n2) {
    float2 v = *(const float2*)(in + 2 * i);
    ((unsigned*)out)[i] = pack2bf(v.x, v.y);
  }
}

// ---------------- tiled WMMA GEMM ----------------
// C[M=4096,N] = A_bf16[M,K] * B_f32[K,N] (+bias)(+resid)(relu?) -> outF/outB
// Block tile 128(M) x 64(N), K-step 64. 8 waves; each wave owns a 32x32 C tile
// -> 8 v_wmma per K-step per wave with full A/B fragment reuse.
// A tile moved with async global->LDS (no VGPR round trip); B tile converted
// f32->bf16 in transit and stored transposed [n][k] with packed b64 stores.
__global__ __launch_bounds__(256)
void gemm_bf16_kernel(const bf16* __restrict__ A, int lda,
                      const float* __restrict__ B, int ldbk, long ldblk,
                      const float* __restrict__ bias,
                      const float* __restrict__ resid,
                      float* __restrict__ outF, bf16* __restrict__ outB,
                      int N, int K, int relu) {
  __shared__ __align__(16) bf16 sA[128][64];  // [m][k]  16 KB
  __shared__ __align__(16) bf16 sB[64][72];   // [n][k]  transposed, 144B rows

  const int tid = threadIdx.x;
  const int wave = tid >> 5;
  const int wm = wave & 3;          // 4 wave-rows  (32 M each)
  const int wn = wave >> 2;         // 2 wave-cols  (32 N each)
  const int lane = tid & 31;
  const int g = lane >> 4;
  const int nl = lane & 15;
  const int m0 = blockIdx.y * 128;
  const int n0 = blockIdx.x * 64;
  const float* Bt = B + (long)(n0 >> 6) * ldblk;

  v8f c00 = v8f_zero(), c01 = v8f_zero(), c10 = v8f_zero(), c11 = v8f_zero();

  // A-tile copy role: 128 rows x 64 cols bf16; each thread moves 64B
  const int ar = tid >> 1;
  const int ah = (tid & 1) * 32;
  // B-tile role: each thread converts a 4(k) x 4(n) f32 micro-tile
  const int bk = (tid >> 4) * 4;        // 0,4,..,60
  const int bn = (tid & 15) * 4;        // 0,4,..,60

  for (int k0 = 0; k0 < K; k0 += 64) {
    { // A tile: 4x async b128 per thread (ASYNCcnt path)
      const bf16* src = A + (long)(m0 + ar) * lda + k0 + ah;
      bf16* dst = &sA[ar][ah];
      async_cp16(dst,      src);
      async_cp16(dst + 8,  src + 8);
      async_cp16(dst + 16, src + 16);
      async_cp16(dst + 24, src + 24);
    }
    { // B tile: 4 rows of float4, packed into 4x ds_store_b64 (transposed)
      const float* bp = Bt + (long)(k0 + bk) * ldbk + bn;
      float4 w0 = *(const float4*)(bp);
      float4 w1 = *(const float4*)(bp + ldbk);
      float4 w2 = *(const float4*)(bp + 2 * (long)ldbk);
      float4 w3 = *(const float4*)(bp + 3 * (long)ldbk);
      uint2 t;
      t.x = pack2bf(w0.x, w1.x); t.y = pack2bf(w2.x, w3.x);
      *(uint2*)&sB[bn + 0][bk] = t;
      t.x = pack2bf(w0.y, w1.y); t.y = pack2bf(w2.y, w3.y);
      *(uint2*)&sB[bn + 1][bk] = t;
      t.x = pack2bf(w0.z, w1.z); t.y = pack2bf(w2.z, w3.z);
      *(uint2*)&sB[bn + 2][bk] = t;
      t.x = pack2bf(w0.w, w1.w); t.y = pack2bf(w2.w, w3.w);
      *(uint2*)&sB[bn + 3][bk] = t;
    }
    if (k0 + 64 < K)  // gfx1250 global_prefetch for next B tile
      __builtin_prefetch(Bt + (long)(k0 + 64 + bk) * ldbk + bn, 0, 0);
    wait_async0();          // A tile resident in LDS
    __syncthreads();

#pragma unroll
    for (int kk = 0; kk < 64; kk += 32) {
      v16bf a0 = load_frag_a(&sA[0][0], 64, wm * 32,      kk);
      v16bf a1 = load_frag_a(&sA[0][0], 64, wm * 32 + 16, kk);
      v16bf b0 = load_frag_b(&sB[0][0], 72, wn * 32,      kk);
      v16bf b1 = load_frag_b(&sB[0][0], 72, wn * 32 + 16, kk);
      c00 = wmma_bf16(a0, b0, c00);
      c01 = wmma_bf16(a0, b1, c01);
      c10 = wmma_bf16(a1, b0, c10);
      c11 = wmma_bf16(a1, b1, c11);
    }
    __syncthreads();
  }

  // epilogue: C-layout row = r + 8*g within each 16-row tile, col = nl
#pragma unroll
  for (int mi = 0; mi < 2; ++mi) {
    v8f cl = mi ? c10 : c00;
    v8f ch = mi ? c11 : c01;
#pragma unroll
    for (int r = 0; r < 8; ++r) {
      int row  = m0 + wm * 32 + mi * 16 + r + 8 * g;
      int col0 = n0 + wn * 32 + nl;
      int col1 = col0 + 16;
      float v0 = cl[r], v1 = ch[r];
      if (bias)  { v0 += bias[col0];                  v1 += bias[col1]; }
      if (resid) { v0 += resid[(long)row * N + col0]; v1 += resid[(long)row * N + col1]; }
      if (relu)  { v0 = fmaxf(v0, 0.f);               v1 = fmaxf(v1, 0.f); }
      if (outF)  { outF[(long)row * N + col0] = v0;   outF[(long)row * N + col1] = v1; }
      if (outB)  { outB[(long)row * N + col0] = f2bf(v0);
                   outB[(long)row * N + col1] = f2bf(v1); }
    }
  }
}

// ---------------- flash attention (per b, head, 64-row Q block) ----------------
__global__ __launch_bounds__(128)
void flash_attn_kernel(const bf16* __restrict__ Q, const bf16* __restrict__ K,
                       const bf16* __restrict__ V, bf16* __restrict__ O) {
  __shared__ __align__(16) bf16 sQ[64][64];
  __shared__ __align__(16) bf16 sK[64][64];   // [s][d]  -> B layout for S = Q K^T
  __shared__ __align__(16) bf16 sVt[64][72];  // [d][s]  -> B layout for P V
  __shared__ __align__(16) bf16 sP[64][72];   // staging P into A layout

  const int tid = threadIdx.x;
  const int wave = tid >> 5;       // 4 waves x 16 Q rows
  const int lane = tid & 31;
  const int g = lane >> 4;
  const int nl = lane & 15;

  const int qt0 = blockIdx.x * 64;
  const int b = blockIdx.y / NH;
  const int h = blockIdx.y % NH;
  const long rowbase = (long)(b * SEQ) * E_DIM + h * HS;

  const int lr = tid >> 1, lh = (tid & 1) * 32;   // tile-copy role

  // load 64x64 Q tile (async global->LDS)
  {
    const bf16* src = Q + rowbase + (long)(qt0 + lr) * E_DIM + lh;
    bf16* dst = &sQ[lr][lh];
    async_cp16(dst,      src);
    async_cp16(dst + 8,  src + 8);
    async_cp16(dst + 16, src + 16);
    async_cp16(dst + 24, src + 24);
  }

  float mrow[8], lrow[8];
  v8f o[4];
#pragma unroll
  for (int r = 0; r < 8; ++r) { mrow[r] = -1e30f; lrow[r] = 0.f; }
#pragma unroll
  for (int dt = 0; dt < 4; ++dt) o[dt] = v8f_zero();

  for (int s0 = 0; s0 <= qt0; s0 += 64) {   // causal: only blocks at/below diagonal
    { // K tile async, V tile transposed through VGPRs
      const bf16* ksrc = K + rowbase + (long)(s0 + lr) * E_DIM + lh;
      bf16* kdst = &sK[lr][lh];
      async_cp16(kdst,      ksrc);
      async_cp16(kdst + 8,  ksrc + 8);
      async_cp16(kdst + 16, ksrc + 16);
      async_cp16(kdst + 24, ksrc + 24);
      const bf16* vs = V + rowbase + (long)(s0 + lr) * E_DIM + lh;
#pragma unroll
      for (int j = 0; j < 32; ++j) sVt[lh + j][lr] = vs[j];   // transpose V
    }
    wait_async0();          // Q (first pass) + K resident
    __syncthreads();

    // S = Q K^T : wave rows [wave*16, +16), all 64 s-columns
    v8f sacc[4];
#pragma unroll
    for (int st = 0; st < 4; ++st) sacc[st] = v8f_zero();
#pragma unroll
    for (int kb = 0; kb < 64; kb += 32) {
      v16bf aq = load_frag_a(&sQ[0][0], 64, wave * 16, kb);
#pragma unroll
      for (int st = 0; st < 4; ++st) {
        v16bf bk2 = load_frag_b(&sK[0][0], 64, st * 16, kb);
        sacc[st] = wmma_bf16(aq, bk2, sacc[st]);
      }
    }

    const bool diag = (s0 == qt0);
#pragma unroll
    for (int st = 0; st < 4; ++st)
#pragma unroll
      for (int r = 0; r < 8; ++r) {
        float vv = sacc[st][r] * 0.03125f;  // 1/sqrt(E) per reference
        if (diag && (s0 + st * 16 + nl) > (qt0 + wave * 16 + r + 8 * g)) vv = -1e30f;
        sacc[st][r] = vv;
      }

    // online softmax; row stats reduced across the 16-lane half-wave
    float alpha[8];
#pragma unroll
    for (int r = 0; r < 8; ++r) {
      float mx = sacc[0][r];
#pragma unroll
      for (int st = 1; st < 4; ++st) mx = fmaxf(mx, sacc[st][r]);
#pragma unroll
      for (int m = 1; m < 16; m <<= 1) mx = fmaxf(mx, __shfl_xor(mx, m, 32));
      float mn = fmaxf(mrow[r], mx);
      alpha[r] = __expf(mrow[r] - mn);
      mrow[r] = mn;
      float rs = 0.f;
#pragma unroll
      for (int st = 0; st < 4; st += 2) {
        float p0 = __expf(sacc[st][r] - mn);
        float p1 = __expf(sacc[st + 1][r] - mn);
        rs += p0 + p1;
        // same-wave DS producer/consumer (DS ops are in-order within a wave)
        sP[wave * 16 + r + 8 * g][st * 16 + nl]       = f2bf(p0);
        sP[wave * 16 + r + 8 * g][(st + 1) * 16 + nl] = f2bf(p1);
      }
#pragma unroll
      for (int m = 1; m < 16; m <<= 1) rs += __shfl_xor(rs, m, 32);
      lrow[r] = lrow[r] * alpha[r] + rs;
    }
#pragma unroll
    for (int dt = 0; dt < 4; ++dt)
#pragma unroll
      for (int r = 0; r < 8; ++r) o[dt][r] *= alpha[r];

    // O += P V
#pragma unroll
    for (int kb = 0; kb < 64; kb += 32) {
      v16bf ap = load_frag_a(&sP[0][0], 72, wave * 16, kb);
#pragma unroll
      for (int dt = 0; dt < 4; ++dt) {
        v16bf bv = load_frag_b(&sVt[0][0], 72, dt * 16, kb);
        o[dt] = wmma_bf16(ap, bv, o[dt]);
      }
    }
    __syncthreads();   // protect sK/sVt for next s-block
  }

#pragma unroll
  for (int r = 0; r < 8; ++r) {
    float inv = 1.f / lrow[r];
    int t = qt0 + wave * 16 + r + 8 * g;
#pragma unroll
    for (int dt = 0; dt < 4; ++dt)
      O[rowbase + (long)t * E_DIM + dt * 16 + nl] = f2bf(o[dt][r] * inv);
  }
}

// ---------------- host orchestration ----------------
extern "C" void kernel_launch(void* const* d_in, const int* in_sizes, int n_in,
                              void* d_out, int out_size, void* d_ws, size_t ws_size,
                              hipStream_t stream) {
  const int*   x       = (const int*)  d_in[0];
  const float* tok_emb = (const float*)d_in[1];
  const float* pos_emb = (const float*)d_in[2];
  const float* Wq      = (const float*)d_in[3];
  const float* Wk      = (const float*)d_in[4];
  const float* Wv      = (const float*)d_in[5];
  const float* Wo      = (const float*)d_in[6];
  const float* bo      = (const float*)d_in[7];
  const float* ln1_g   = (const float*)d_in[8];
  const float* ln1_b   = (const float*)d_in[9];
  const float* ln2_g   = (const float*)d_in[10];
  const float* ln2_b   = (const float*)d_in[11];
  const float* W1      = (const float*)d_in[12];
  const float* b1      = (const float*)d_in[13];
  const float* W2      = (const float*)d_in[14];
  const float* b2      = (const float*)d_in[15];
  const float* fW1     = (const float*)d_in[16];
  const float* fb1     = (const float*)d_in[17];
  const float* fW2     = (const float*)d_in[18];
  const float* fb2     = (const float*)d_in[19];
  const float* head_W  = (const float*)d_in[20];
  const float* head_b  = (const float*)d_in[21];
  float* out = (float*)d_out;

  // workspace carve-up (~88 MB)
  char* p = (char*)d_ws;
  float* z  = (float*)p; p += (size_t)MTOT * E_DIM * sizeof(float);
  bf16*  h  = (bf16*)p;  p += (size_t)MTOT * E_DIM * sizeof(bf16);
  bf16*  qb = (bf16*)p;  p += (size_t)MTOT * E_DIM * sizeof(bf16);
  bf16*  kb = (bf16*)p;  p += (size_t)MTOT * E_DIM * sizeof(bf16);
  bf16*  vb = (bf16*)p;  p += (size_t)MTOT * E_DIM * sizeof(bf16);
  bf16*  ab = (bf16*)p;  p += (size_t)MTOT * E_DIM * sizeof(bf16);
  bf16*  ff = (bf16*)p;  p += (size_t)MTOT * FF_DIM * sizeof(bf16);

  dim3 blk256(256), blk128(128);
  dim3 gRow(MTOT);
  dim3 gG1(E_DIM / 64, MTOT / 128);    // N = 1024
  dim3 gG4(FF_DIM / 64, MTOT / 128);   // N = 4096
  dim3 gGV(VOCAB / 64, MTOT / 128);    // N = 32000
  dim3 gAtt(SEQ / 64, BATCH * NH);
  dim3 gCvt((MTOT * E_DIM / 2) / 256);

  embed_kernel<<<gRow, blk256, 0, stream>>>(x, tok_emb, pos_emb, z);

  for (int l = 0; l < NLAYER; ++l) {
    const float* wq = Wq + (long)l * NH * E_DIM * HS;
    const float* wk = Wk + (long)l * NH * E_DIM * HS;
    const float* wv = Wv + (long)l * NH * E_DIM * HS;
    const float* wo = Wo + (long)l * E_DIM * E_DIM;
    const float* w1 = W1 + (long)l * E_DIM * FF_DIM;
    const float* w2 = W2 + (long)l * FF_DIM * E_DIM;

    layernorm_kernel<<<gRow, blk256, 0, stream>>>(z, ln1_g + l * E_DIM, ln1_b + l * E_DIM, h);
    // QKV: Wq[l] is [H][E][HS]; per-64-col block stride = E*HS, k stride = HS
    gemm_bf16_kernel<<<gG1, blk256, 0, stream>>>(h, E_DIM, wq, HS, (long)E_DIM * HS,
                                                 nullptr, nullptr, nullptr, qb, E_DIM, E_DIM, 0);
    gemm_bf16_kernel<<<gG1, blk256, 0, stream>>>(h, E_DIM, wk, HS, (long)E_DIM * HS,
                                                 nullptr, nullptr, nullptr, kb, E_DIM, E_DIM, 0);
    gemm_bf16_kernel<<<gG1, blk256, 0, stream>>>(h, E_DIM, wv, HS, (long)E_DIM * HS,
                                                 nullptr, nullptr, nullptr, vb, E_DIM, E_DIM, 0);

    flash_attn_kernel<<<gAtt, blk128, 0, stream>>>(qb, kb, vb, ab);

    // z = z + a @ Wo + bo
    gemm_bf16_kernel<<<gG1, blk256, 0, stream>>>(ab, E_DIM, wo, E_DIM, 64,
                                                 bo + l * E_DIM, z, z, nullptr, E_DIM, E_DIM, 0);
    layernorm_kernel<<<gRow, blk256, 0, stream>>>(z, ln2_g + l * E_DIM, ln2_b + l * E_DIM, h);
    // ff = relu(h @ W1 + b1)
    gemm_bf16_kernel<<<gG4, blk256, 0, stream>>>(h, E_DIM, w1, FF_DIM, 64,
                                                 b1 + l * FF_DIM, nullptr, nullptr, ff,
                                                 FF_DIM, E_DIM, 1);
    // z = z + ff @ W2 + b2
    gemm_bf16_kernel<<<gG1, blk256, 0, stream>>>(ff, FF_DIM, w2, E_DIM, 64,
                                                 b2 + l * E_DIM, z, z, nullptr, E_DIM, FF_DIM, 0);
  }

  // non-residual final feedforward: z = relu(z @ fW1 + fb1) @ fW2 + fb2
  cvt_f32_bf16_kernel<<<gCvt, blk256, 0, stream>>>(z, h, MTOT * E_DIM / 2);
  gemm_bf16_kernel<<<gG4, blk256, 0, stream>>>(h, E_DIM, fW1, FF_DIM, 64,
                                               fb1, nullptr, nullptr, ff, FF_DIM, E_DIM, 1);
  gemm_bf16_kernel<<<gG1, blk256, 0, stream>>>(ff, FF_DIM, fW2, E_DIM, 64,
                                               fb2, nullptr, z, nullptr, E_DIM, FF_DIM, 0);
  // logits = z @ head_W + head_b
  cvt_f32_bf16_kernel<<<gCvt, blk256, 0, stream>>>(z, h, MTOT * E_DIM / 2);
  gemm_bf16_kernel<<<gGV, blk256, 0, stream>>>(h, E_DIM, head_W, VOCAB, 64,
                                               head_b, nullptr, out, nullptr, VOCAB, E_DIM, 0);

  (void)in_sizes; (void)n_in; (void)out_size; (void)ws_size;
}
